// FeedForwardQuantum_65481071402794
// MI455X (gfx1250) — matrix operand order repaired and checked by reading, not compiled, not run on hardware
//
#include <hip/hip_runtime.h>
#include <stdint.h>

// Problem dims (from reference)
#define EMBED  1024
#define FFN    4096
#define NW     8
#define TOKENS (16 * 2048)

// Tile config: block = 256 threads = 8 waves (wave32)
#define TM 128   // tokens per block
#define TN 256   // embed cols per block
#define TK 32    // K chunk (matches wmma_f32_16x16x32_bf16)

#define A_STRIDE 40   // bf16 elems per LDS row (80B: 16B-aligned, conflict-spreading)
#define B_STRIDE 40

typedef __bf16 bf16_t;
typedef bf16_t v16bf __attribute__((ext_vector_type(16)));
typedef float  v8f   __attribute__((ext_vector_type(8)));

union FragU { uint4 q[2]; v16bf v; };

// round-to-nearest-even f32 -> bf16, packed pair
static __device__ __forceinline__ uint32_t pack_bf16x2(float a, float b) {
    uint32_t ua = __float_as_uint(a);
    uint32_t ub = __float_as_uint(b);
    ua = (ua + 0x7FFFu + ((ua >> 16) & 1u)) >> 16;
    ub = (ub + 0x7FFFu + ((ub >> 16) & 1u)) >> 16;
    return (ub << 16) | (ua & 0xFFFFu);
}

// ---------------------------------------------------------------------------
// Prep kernel: W2 [1024][4096] f32  ->  W2bf blocked [kc=128][n=1024][kk=32] bf16
// Each main-loop B tile (given kc, Nbase) becomes one contiguous 16 KB span.
// ---------------------------------------------------------------------------
__global__ __launch_bounds__(256)
void w2_pack(const float* __restrict__ W2, uint16_t* __restrict__ W2bf) {
    const int g    = blockIdx.x * 256 + threadIdx.x;  // 0 .. 4096*1024/8 - 1
    const int part = g & 3;                           // 8-elem slice of a 32-elem chunk
    const int nc   = g >> 2;                          // kc*1024 + n
    const int n    = nc & (EMBED - 1);
    const int kc   = nc >> 10;
    const float* src = W2 + (size_t)n * FFN + kc * TK + part * 8;
    const float4 f0 = *(const float4*)(src);
    const float4 f1 = *(const float4*)(src + 4);
    uint4 o;
    o.x = pack_bf16x2(f0.x, f0.y);
    o.y = pack_bf16x2(f0.z, f0.w);
    o.z = pack_bf16x2(f1.x, f1.y);
    o.w = pack_bf16x2(f1.z, f1.w);
    *(uint4*)(W2bf + (size_t)nc * TK + part * 8) = o;
}

// ---------------------------------------------------------------------------
// Fused FFN kernel.
// PRECONV=true : B tile streamed bf16 global->LDS via global_load_async_to_lds_b128
// PRECONV=false: B tile converted f32->bf16 in-loop (fallback, no workspace)
// ---------------------------------------------------------------------------
template <bool PRECONV>
__global__ __launch_bounds__(256)
void ffq_fused(const float* __restrict__ x,          // [16,2048,1024]
               const float* __restrict__ theta,      // [8]
               const float* __restrict__ W1,         // [4096, 8]
               const float* __restrict__ b1,         // [4096]
               const float* __restrict__ W2,         // [1024, 4096]
               const uint16_t* __restrict__ W2bf,    // blocked bf16 (PRECONV)
               const float* __restrict__ b2,         // [1024]
               float* __restrict__ out)              // [16,2048,1024]
{
    __shared__ float                 qzs[TM][NW];           // 4 KB
    __shared__ __align__(16) uint16_t Alds[TM][A_STRIDE];   // 10 KB  h tile (bf16)
    __shared__ __align__(16) uint16_t Blds[TN][B_STRIDE];   // 20 KB  W2 tile (bf16)

    const int tid   = threadIdx.x;
    const int Nbase = blockIdx.x * TN;   // 0..768
    const int Mbase = blockIdx.y * TM;   // 0..32640

    // ---- preamble: qz[m][w] = cos(x[token, w]) * cos(theta[w]) ----
    {
        const int m  = tid >> 1;
        const int wh = (tid & 1) * 4;
        const size_t xoff = (size_t)(Mbase + m) * EMBED + wh;
        const float4 xv = *(const float4*)(x + xoff);
        const float4 tv = *(const float4*)(theta + wh);
        float4 q;
        q.x = cosf(xv.x) * cosf(tv.x);
        q.y = cosf(xv.y) * cosf(tv.y);
        q.z = cosf(xv.z) * cosf(tv.z);
        q.w = cosf(xv.w) * cosf(tv.w);
        *(float4*)&qzs[m][wh] = q;
    }

    // wave decomposition: 2 (M) x 4 (N) waves, each owns a 64x64 output region
    const int lane   = tid & 31;
    const int wid    = tid >> 5;
    const int waveM  = wid & 1;   // 0..1
    const int waveN  = wid >> 1;  // 0..3
    const int lane16 = lane & 15;
    const int laneHi = lane >> 4; // 0/1

    v8f acc[4][4] = {};           // 16 tiles of 16x16 f32 = 128 VGPRs

    // A-generation mapping: thread owns 2 adjacent k columns x 8 token rows
    const int kk2   = (tid & 15) * 2;
    const int mbase = (tid >> 4) * 8;

    // LDS byte offset of Blds (generic-pointer low 32 bits == LDS address)
    const uint32_t BldsOff = (uint32_t)(uintptr_t)&Blds[0][0];

    __syncthreads();

    for (int k0 = 0; k0 < FFN; k0 += TK) {
        // ---- B tile: W2 cols [k0..k0+31] for rows [Nbase..Nbase+255] ----
        if constexpr (PRECONV) {
            // blocked bf16: tile is 16 KB contiguous at ((k0/TK)*1024 + Nbase)*TK
            const uint16_t* w2base =
                W2bf + ((size_t)(k0 / TK) * EMBED + Nbase) * TK;
#pragma unroll
            for (int i = 0; i < 4; ++i) {
                const int seg = tid + i * 256;         // 16B segment, 0..1023
                const int row = seg >> 2;              // n row within tile
                const int s4  = seg & 3;               // 16B slot within row
                const uint32_t ldsoff =
                    BldsOff + (uint32_t)(row * (B_STRIDE * 2) + s4 * 16);
                const uint64_t ga =
                    (uint64_t)(uintptr_t)w2base + (uint64_t)seg * 16;
                asm volatile("global_load_async_to_lds_b128 %0, %1, off"
                             :: "v"(ldsoff), "v"(ga) : "memory");
            }
            if (k0 + TK < FFN)
                __builtin_prefetch((const char*)w2base + TN * TK * 2, 0, 0);
        } else {
            const float* w2r = W2 + (size_t)(Nbase + tid) * FFN + k0;
            uint32_t p[16];
#pragma unroll
            for (int i = 0; i < 8; ++i) {
                const float4 f = *(const float4*)(w2r + i * 4);
                p[2 * i]     = pack_bf16x2(f.x, f.y);
                p[2 * i + 1] = pack_bf16x2(f.z, f.w);
            }
            uint4* dst = (uint4*)&Blds[tid][0];
#pragma unroll
            for (int i = 0; i < 4; ++i)
                dst[i] = make_uint4(p[4*i], p[4*i+1], p[4*i+2], p[4*i+3]);
        }

        // ---- produce A tile: h[128][32] = relu(qz @ W1' + b1), bf16 in LDS ----
        // (overlaps with the in-flight async B copy)
        {
            const float* w1r = W1 + (size_t)(k0 + kk2) * NW;
            const float4 wa0 = *(const float4*)(w1r);
            const float4 wa1 = *(const float4*)(w1r + 4);
            const float4 wb0 = *(const float4*)(w1r + 8);
            const float4 wb1 = *(const float4*)(w1r + 12);
            const float ba = b1[k0 + kk2];
            const float bb = b1[k0 + kk2 + 1];
#pragma unroll
            for (int mm = 0; mm < 8; ++mm) {
                const float4 q0 = *(const float4*)&qzs[mbase + mm][0];
                const float4 q1 = *(const float4*)&qzs[mbase + mm][4];
                float h0 = ba, h1 = bb;
                h0 = fmaf(q0.x, wa0.x, h0); h1 = fmaf(q0.x, wb0.x, h1);
                h0 = fmaf(q0.y, wa0.y, h0); h1 = fmaf(q0.y, wb0.y, h1);
                h0 = fmaf(q0.z, wa0.z, h0); h1 = fmaf(q0.z, wb0.z, h1);
                h0 = fmaf(q0.w, wa0.w, h0); h1 = fmaf(q0.w, wb0.w, h1);
                h0 = fmaf(q1.x, wa1.x, h0); h1 = fmaf(q1.x, wb1.x, h1);
                h0 = fmaf(q1.y, wa1.y, h0); h1 = fmaf(q1.y, wb1.y, h1);
                h0 = fmaf(q1.z, wa1.z, h0); h1 = fmaf(q1.z, wb1.z, h1);
                h0 = fmaf(q1.w, wa1.w, h0); h1 = fmaf(q1.w, wb1.w, h1);
                h0 = fmaxf(h0, 0.0f);
                h1 = fmaxf(h1, 0.0f);
                *(uint32_t*)&Alds[mbase + mm][kk2] = pack_bf16x2(h0, h1);
            }
        }

        if constexpr (PRECONV) {
            // all 4 async B transfers must have landed in LDS before the barrier
            asm volatile("s_wait_asynccnt 0x0" ::: "memory");
        }
        __syncthreads();

        // ---- fragments + 16 WMMAs per wave ----
        // A 16-bit layout: lane<16 holds row M=lane16, K 0..7 & 16..23;
        //                  lane>=16 holds K 8..15 & 24..31.
        FragU afr[4];
#pragma unroll
        for (int mi = 0; mi < 4; ++mi) {
            const uint16_t* pa = &Alds[waveM * 64 + mi * 16 + lane16][laneHi * 8];
            afr[mi].q[0] = *(const uint4*)(pa);
            afr[mi].q[1] = *(const uint4*)(pa + 16);
        }
        // B 16-bit layout: lane<16 holds col N=lane16, K 0..7 & 16..23;
        //                  lane>=16 holds K 8..15 & 24..31.
#pragma unroll
        for (int ni = 0; ni < 4; ++ni) {
            FragU bfr;
            const uint16_t* pb = &Blds[waveN * 64 + ni * 16 + lane16][laneHi * 8];
            bfr.q[0] = *(const uint4*)(pb);
            bfr.q[1] = *(const uint4*)(pb + 16);
#pragma unroll
            for (int mi = 0; mi < 4; ++mi) {
                acc[mi][ni] = __builtin_amdgcn_wmma_f32_16x16x32_bf16(
                    false, afr[mi].v, false, bfr.v,
                    (short)0, acc[mi][ni], false, false);
            }
        }

        __syncthreads();
    }

    // ---- epilogue: + b2, store f32 ----
    // C layout: lane<16 -> col N=lane16, VGPR r -> row M=r; lane>=16 -> row M=r+8
#pragma unroll
    for (int ni = 0; ni < 4; ++ni) {
        const int n = Nbase + waveN * 64 + ni * 16 + lane16;
        const float bias = b2[n];
#pragma unroll
        for (int mi = 0; mi < 4; ++mi) {
            const int mrow = Mbase + waveM * 64 + mi * 16 + laneHi * 8;
#pragma unroll
            for (int r = 0; r < 8; ++r) {
                out[(size_t)(mrow + r) * EMBED + n] = acc[mi][ni][r] + bias;
            }
        }
    }
    (void)W2; (void)W2bf;
}

extern "C" void kernel_launch(void* const* d_in, const int* in_sizes, int n_in,
                              void* d_out, int out_size, void* d_ws, size_t ws_size,
                              hipStream_t stream) {
    (void)in_sizes; (void)n_in; (void)out_size;
    const float* x     = (const float*)d_in[0];
    const float* theta = (const float*)d_in[1];
    const float* W1    = (const float*)d_in[2];
    const float* b1    = (const float*)d_in[3];
    const float* W2    = (const float*)d_in[4];
    const float* b2    = (const float*)d_in[5];
    float* out = (float*)d_out;

    const dim3 grid(EMBED / TN, TOKENS / TM);   // (4, 256) workgroups
    const size_t need = (size_t)FFN * EMBED * sizeof(uint16_t);  // 8 MB

    if (ws_size >= need && d_ws != nullptr) {
        uint16_t* W2bf = (uint16_t*)d_ws;
        w2_pack<<<dim3((FFN * EMBED / 8) / 256), 256, 0, stream>>>(W2, W2bf);
        ffq_fused<true><<<grid, 256, 0, stream>>>(x, theta, W1, b1, W2, W2bf, b2, out);
    } else {
        ffq_fused<false><<<grid, 256, 0, stream>>>(x, theta, W1, b1, W2, nullptr, b2, out);
    }
}
